// DMN_34978213658644
// MI455X (gfx1250) — matrix operand order repaired and checked by reading, not compiled, not run on hardware
//
#include <hip/hip_runtime.h>
#include <cstdint>
#include <cstddef>

// ---------------- WMMA vector types (gfx1250, wave32) ----------------
typedef __attribute__((ext_vector_type(16))) __bf16 v16bf;
typedef __attribute__((ext_vector_type(8)))  __bf16 v8bf;
typedef __attribute__((ext_vector_type(8)))  float  v8f;

// ---------------- problem dims ----------------
#define BB     128
#define T_INFO 1024
#define NSENT  64
#define T_Q    64
#define EMB    50
#define RU     400
#define MU     300
#define VU     2000

// padded dims (K mult of 32, N mult of 16)
#define EP    64
#define HP_R  416
#define HP_M  320
#define HP_A  2016
#define NT_R  (HP_R/16)   // 26
#define NT_M  (HP_M/16)   // 20
#define NT_A  (HP_A/16)   // 126
#define KC_E  (EP/32)     // 2
#define KC_R  (HP_R/32)   // 13
#define KC_M  (HP_M/32)   // 10
#define KC_A  (HP_A/32)   // 63

// ---------------- device helpers ----------------
__device__ __forceinline__ float sigm(float x) { return 1.0f / (1.0f + __expf(-x)); }

// A-fragment: 16 rows x 32 K of row-major bf16 matrix, per CDNA5 16-bit A layout.
// lane<16: K = k0+[0..7], k0+16+[0..7]; lane>=16: K = k0+8+[0..7], k0+24+[0..7]
__device__ __forceinline__ v16bf loadA(const __bf16* __restrict__ base, int ld, int row0, int k0) {
  int lane = threadIdx.x & 31;
  const __bf16* rp = base + (size_t)(row0 + (lane & 15)) * ld + k0 + ((lane >= 16) ? 8 : 0);
  union { v16bf v; v8bf h[2]; } u;
  u.h[0] = *(const v8bf*)rp;
  u.h[1] = *(const v8bf*)(rp + 16);
  return u.v;
}

// B-fragment from fragment-major packed weights: 512 halves per (tile,kc) fragment,
// lane-contiguous 32B loads.
__device__ __forceinline__ v16bf loadB(const __bf16* __restrict__ pack, int KC, int tile, int kc) {
  int lane = threadIdx.x & 31;
  return *(const v16bf*)(pack + (((size_t)tile * KC + kc) << 9) + (lane << 4));
}

#define WMMA_BF16(A, Bm, C) \
  __builtin_amdgcn_wmma_f32_16x16x32_bf16(false, (A), false, (Bm), (short)0, (C), false, false)

// device-wide sense-reversing barrier (persistent-kernel grid sync)
__device__ __forceinline__ void gbar(unsigned* bar) {
  __syncthreads();
  if (threadIdx.x == 0) {
    __threadfence();
    volatile unsigned* cnt = bar;
    volatile unsigned* gen = bar + 1;
    unsigned g = *gen;
    if (atomicAdd((unsigned*)bar, 1u) == gridDim.x - 1) {
      *cnt = 0;
      __threadfence();
      atomicAdd((unsigned*)(bar + 1), 1u);
    } else {
      while (*gen == g) __builtin_amdgcn_s_sleep(1);
    }
  }
  __syncthreads();
}

// ---------------- utility kernels ----------------
__global__ void k_fill_u32(uint32_t* p, uint32_t v, size_t n) {
  size_t i  = (size_t)blockIdx.x * blockDim.x + threadIdx.x;
  size_t st = (size_t)gridDim.x * blockDim.x;
  for (; i < n; i += st) p[i] = v;
}

// pack f32 weight [Hin, 3*Hv] -> bf16 fragment-major [tiles over 3*Hp cols][KC][512]
// (transposed-for-B, gate-blocked with zero padding)
__global__ void k_pack(const float* __restrict__ src, int Hin, int Hv,
                       int KC, int NT3, int Hp, __bf16* __restrict__ dst) {
  size_t total = (size_t)KC * NT3 * 512;
  size_t i  = (size_t)blockIdx.x * blockDim.x + threadIdx.x;
  size_t st = (size_t)gridDim.x * blockDim.x;
  for (; i < total; i += st) {
    size_t f = i >> 9;
    int idx  = (int)(i & 511);
    int lane = idx >> 4, e = idx & 15;
    int kc   = (int)(f % KC);
    int tile = (int)(f / KC);
    int npad = tile * 16 + (lane & 15);
    int g = npad / Hp, c = npad % Hp;
    int k = kc * 32 + ((lane >> 4) << 3) + ((e < 8) ? e : 8 + e);
    float v = (c < Hv && k < Hin) ? src[(size_t)k * (3 * Hv) + (size_t)g * Hv + c] : 0.0f;
    dst[i] = (__bf16)v;
  }
}

// gather embeddings -> bf16 padded x buffer [(b*T+t)*Ep + e]
__global__ void k_embed(const int* __restrict__ ids, const float* __restrict__ emb,
                        int T, int E, int Ep, __bf16* __restrict__ dst) {
  size_t total = (size_t)BB * T * Ep;
  size_t i  = (size_t)blockIdx.x * blockDim.x + threadIdx.x;
  size_t st = (size_t)gridDim.x * blockDim.x;
  for (; i < total; i += st) {
    int e = (int)(i % Ep);
    size_t bt = i / Ep;
    int id = ids[bt];
    dst[i] = (e < E) ? (__bf16)emb[(size_t)id * E + e] : (__bf16)0.0f;
  }
}

// scatter chain: for each (b,t) a linked list of j with info_idx[b,j]==t
__global__ void k_build_match(const int* __restrict__ info_idx, int T,
                              int* __restrict__ head, int* __restrict__ nxt) {
  int b = blockIdx.x * blockDim.x + threadIdx.x;
  if (b < BB) {
    for (int j = 0; j < NSENT; ++j) {
      int t = info_idx[b * NSENT + j];
      if (t < 0) t = 0;
      if (t >= T) t = T - 1;
      nxt[b * NSENT + j] = head[(size_t)b * T + t];
      head[(size_t)b * T + t] = j;
    }
  }
}

__global__ void k_qtgt(const int* __restrict__ qidx, int* __restrict__ tgt) {
  int b = blockIdx.x * blockDim.x + threadIdx.x;
  if (b < BB) tgt[b] = qidx[b] - 1;
}

// prod = enc_info * pad * enc_q  -> bf16 x-input of memory GRU
__global__ void k_prod(const float* __restrict__ enc_i, const float* __restrict__ enc_q,
                       const int* __restrict__ num_sent, int Hp, int Hv,
                       __bf16* __restrict__ dst) {
  size_t total = (size_t)BB * NSENT * Hp;
  size_t i  = (size_t)blockIdx.x * blockDim.x + threadIdx.x;
  size_t st = (size_t)gridDim.x * blockDim.x;
  for (; i < total; i += st) {
    int c = (int)(i % Hp);
    size_t bj = i / Hp;
    int j = (int)(bj % NSENT);
    int b = (int)(bj / NSENT);
    float m = (j <= num_sent[b]) ? 1.0f : 0.0f;
    float v = (c < Hv) ? enc_i[i] * enc_q[(size_t)b * Hp + c] * m : 0.0f;
    dst[i] = (__bf16)v;
  }
}

// ---------------- persistent GRU scan (tanh candidate) ----------------
// mode 0: scatter rows into enc_out via (head,nxt) chain   [info GRU]
// mode 1: scatter row at t==qtgt[b] into enc_out           [question GRU]
// mode 2: write masked bf16 sequence output                [memory GRU]
__global__ void k_gru_scan_tanh(
    const __bf16* __restrict__ xin, int ldx, int KCx, const __bf16* __restrict__ WP,
    const __bf16* __restrict__ UP, int KCh, int Hp, int Hv, int NTg,
    const float* __restrict__ b0, const float* __restrict__ b1, int T,
    float* __restrict__ hf32, __bf16* __restrict__ hb16,
    int mode, float* __restrict__ enc_out,
    const int* __restrict__ head, const int* __restrict__ nxt,
    const int* __restrict__ qtgt, const int* __restrict__ num_sent,
    __bf16* __restrict__ seq_out, unsigned* bar) {
  int lane = threadIdx.x & 31;
  int wv   = threadIdx.x >> 5;       // 8 waves/block: wave = row tile (16 rows)
  int nt   = blockIdx.x;             // column tile of Hp
  int row0 = wv * 16;
  int colL = nt * 16 + (lane & 15);
  int rbase = row0 + ((lane >= 16) ? 8 : 0);
  int Epx = KCx * 32;

  bool  cv  = colL < Hv;
  float b0z = cv ? b0[colL]          : 0.0f;
  float b0r = cv ? b0[Hv + colL]     : 0.0f;
  float b0h = cv ? b0[2 * Hv + colL] : 0.0f;
  float b1z = cv ? b1[colL]          : 0.0f;
  float b1r = cv ? b1[Hv + colL]     : 0.0f;
  float b1h = cv ? b1[2 * Hv + colL] : 0.0f;

  for (int t = 0; t < T; ++t) {
    int cur = t & 1, nb = cur ^ 1;
    v8f accz = {}, accr = {}, accxh = {}, accuh = {};

    // x_t @ W  (concatenated into same accumulators)
    const __bf16* xa = xin + (size_t)t * Epx;
    for (int kc = 0; kc < KCx; ++kc) {
      v16bf a = loadA(xa, ldx, row0, kc * 32);
      accz  = WMMA_BF16(a, loadB(WP, KCx, 0 * NTg + nt, kc), accz);
      accr  = WMMA_BF16(a, loadB(WP, KCx, 1 * NTg + nt, kc), accr);
      accxh = WMMA_BF16(a, loadB(WP, KCx, 2 * NTg + nt, kc), accxh);
    }
    // h_{t-1} @ U
    const __bf16* hb = hb16 + (size_t)cur * BB * Hp;
    for (int kc = 0; kc < KCh; ++kc) {
      v16bf a = loadA(hb, Hp, row0, kc * 32);
      accz  = WMMA_BF16(a, loadB(UP, KCh, 0 * NTg + nt, kc), accz);
      accr  = WMMA_BF16(a, loadB(UP, KCh, 1 * NTg + nt, kc), accr);
      accuh = WMMA_BF16(a, loadB(UP, KCh, 2 * NTg + nt, kc), accuh);
    }

    const float* hfc = hf32 + (size_t)cur * BB * Hp;
    float*       hfn = hf32 + (size_t)nb  * BB * Hp;
    __bf16*      hbn = hb16 + (size_t)nb  * BB * Hp;
    for (int j = 0; j < 8; ++j) {
      int row = rbase + j;
      float z  = sigm(accz[j] + b0z + b1z);
      float r  = sigm(accr[j] + b0r + b1r);
      float hh = tanhf(accxh[j] + b0h + r * (accuh[j] + b1h));
      float ho = hfc[(size_t)row * Hp + colL];
      float hn = z * ho + (1.0f - z) * hh;
      hfn[(size_t)row * Hp + colL] = hn;
      hbn[(size_t)row * Hp + colL] = (__bf16)hn;
      if (mode == 0) {
        int m = head[(size_t)row * T + t];
        while (m >= 0) {
          enc_out[((size_t)row * NSENT + m) * Hp + colL] = hn;
          m = nxt[row * NSENT + m];
        }
      } else if (mode == 1) {
        if (qtgt[row] == t) enc_out[(size_t)row * Hp + colL] = hn;
      } else {
        float msk = (t <= num_sent[row]) ? 1.0f : 0.0f;
        seq_out[((size_t)row * T + t) * Hp + colL] = (__bf16)(hn * msk);
      }
    }
    gbar(bar);
  }
}

// ---------------- persistent GRU scan (softmax candidate) = answer GRU ----------------
__global__ void k_gru_scan_softmax(
    const __bf16* __restrict__ xin, int ldx, int KCx, const __bf16* __restrict__ WP,
    const __bf16* __restrict__ UP, int KCh, int Hp, int Hv, int NTg,
    const float* __restrict__ b0, const float* __restrict__ b1, int T,
    float* __restrict__ hf32, __bf16* __restrict__ hb16,
    float* __restrict__ sbuf, float* __restrict__ zbuf,
    float* __restrict__ rowmax, float* __restrict__ rowsum,
    float* __restrict__ out, unsigned* bar) {
  int lane = threadIdx.x & 31;
  int wv   = threadIdx.x >> 5;
  int nt   = blockIdx.x;
  int row0 = wv * 16;
  int colL = nt * 16 + (lane & 15);
  int rbase = row0 + ((lane >= 16) ? 8 : 0);
  int Epx = KCx * 32;
  int gw  = blockIdx.x * 8 + wv;   // global wave id (for reduction phase)

  bool  cv  = colL < Hv;
  float b0z = cv ? b0[colL]          : 0.0f;
  float b0r = cv ? b0[Hv + colL]     : 0.0f;
  float b0h = cv ? b0[2 * Hv + colL] : 0.0f;
  float b1z = cv ? b1[colL]          : 0.0f;
  float b1r = cv ? b1[Hv + colL]     : 0.0f;
  float b1h = cv ? b1[2 * Hv + colL] : 0.0f;

  for (int t = 0; t < T; ++t) {
    int cur = t & 1, nb = cur ^ 1;
    // ---- phase 1: rec GEMM + z / pre-softmax s ----
    v8f accz = {}, accr = {}, accxh = {}, accuh = {};
    const __bf16* xa = xin + (size_t)t * Epx;
    for (int kc = 0; kc < KCx; ++kc) {
      v16bf a = loadA(xa, ldx, row0, kc * 32);
      accz  = WMMA_BF16(a, loadB(WP, KCx, 0 * NTg + nt, kc), accz);
      accr  = WMMA_BF16(a, loadB(WP, KCx, 1 * NTg + nt, kc), accr);
      accxh = WMMA_BF16(a, loadB(WP, KCx, 2 * NTg + nt, kc), accxh);
    }
    const __bf16* hb = hb16 + (size_t)cur * BB * Hp;
    for (int kc = 0; kc < KCh; ++kc) {
      v16bf a = loadA(hb, Hp, row0, kc * 32);
      accz  = WMMA_BF16(a, loadB(UP, KCh, 0 * NTg + nt, kc), accz);
      accr  = WMMA_BF16(a, loadB(UP, KCh, 1 * NTg + nt, kc), accr);
      accuh = WMMA_BF16(a, loadB(UP, KCh, 2 * NTg + nt, kc), accuh);
    }
    for (int j = 0; j < 8; ++j) {
      int row = rbase + j;
      float z = sigm(accz[j] + b0z + b1z);
      float r = sigm(accr[j] + b0r + b1r);
      float s = accxh[j] + b0h + r * (accuh[j] + b1h);
      zbuf[(size_t)row * Hp + colL] = z;
      sbuf[(size_t)row * Hp + colL] = s;
    }
    gbar(bar);
    // ---- phase 2: per-row max & sum(exp) over valid V columns ----
    if (gw < BB) {
      int row = gw;
      const float* sr = sbuf + (size_t)row * Hp;
      float mx = -3.0e38f;
      for (int c = lane; c < Hv; c += 32) mx = fmaxf(mx, sr[c]);
      for (int o = 16; o > 0; o >>= 1) mx = fmaxf(mx, __shfl_xor(mx, o, 32));
      float sm = 0.0f;
      for (int c = lane; c < Hv; c += 32) sm += __expf(sr[c] - mx);
      for (int o = 16; o > 0; o >>= 1) sm += __shfl_xor(sm, o, 32);
      if (lane == 0) { rowmax[row] = mx; rowsum[row] = sm; }
    }
    gbar(bar);
    // ---- phase 3: h update + output write ----
    const float* hfc = hf32 + (size_t)cur * BB * Hp;
    float*       hfn = hf32 + (size_t)nb  * BB * Hp;
    __bf16*      hbn = hb16 + (size_t)nb  * BB * Hp;
    for (int j = 0; j < 8; ++j) {
      int row = rbase + j;
      float z  = zbuf[(size_t)row * Hp + colL];
      float s  = sbuf[(size_t)row * Hp + colL];
      float hh = cv ? __expf(s - rowmax[row]) / rowsum[row] : 0.0f;
      float ho = hfc[(size_t)row * Hp + colL];
      float hn = z * ho + (1.0f - z) * hh;
      hfn[(size_t)row * Hp + colL] = hn;
      hbn[(size_t)row * Hp + colL] = (__bf16)hn;
      if (cv) out[((size_t)row * T + t) * Hv + colL] = hn;
    }
    gbar(bar);
  }
}

// ---------------- workspace layout ----------------
static constexpr size_t al(size_t x) { return (x + 255) & ~(size_t)255; }
constexpr size_t N_WIP = (size_t)KC_E * 3 * NT_R * 512;
constexpr size_t N_UIP = (size_t)KC_R * 3 * NT_R * 512;
constexpr size_t N_WMP = (size_t)KC_R * 3 * NT_M * 512;
constexpr size_t N_UMP = (size_t)KC_M * 3 * NT_M * 512;
constexpr size_t N_WAP = (size_t)KC_M * 3 * NT_A * 512;
constexpr size_t N_UAP = (size_t)KC_A * 3 * NT_A * 512;

constexpr size_t O_BAR  = 0;                                        // 256B: barrier slots
constexpr size_t O_HF   = 256;                                      // h f32 ping-pong
constexpr size_t O_HB   = al(O_HF  + (size_t)2 * BB * HP_A * 4);    // h bf16 ping-pong
constexpr size_t O_WIP  = al(O_HB  + (size_t)2 * BB * HP_A * 2);
constexpr size_t O_UIP  = al(O_WIP + N_WIP * 2);
constexpr size_t O_WQP  = al(O_UIP + N_UIP * 2);
constexpr size_t O_UQP  = al(O_WQP + N_WIP * 2);
constexpr size_t O_WMP  = al(O_UQP + N_UIP * 2);
constexpr size_t O_UMP  = al(O_WMP + N_WMP * 2);
constexpr size_t O_WAP  = al(O_UMP + N_UMP * 2);
constexpr size_t O_UAP  = al(O_WAP + N_WAP * 2);
constexpr size_t O_XI   = al(O_UAP + N_UAP * 2);
constexpr size_t O_XQ   = al(O_XI  + (size_t)BB * T_INFO * EP * 2);
constexpr size_t O_EI   = al(O_XQ  + (size_t)BB * T_Q * EP * 2);
constexpr size_t O_EQ   = al(O_EI  + (size_t)BB * NSENT * HP_R * 4);
constexpr size_t O_XM   = al(O_EQ  + (size_t)BB * HP_R * 4);
constexpr size_t O_XA   = al(O_XM  + (size_t)BB * NSENT * HP_R * 2);
constexpr size_t O_SB   = al(O_XA  + (size_t)BB * NSENT * HP_M * 2);
constexpr size_t O_ZB   = al(O_SB  + (size_t)BB * HP_A * 4);
constexpr size_t O_RMX  = al(O_ZB  + (size_t)BB * HP_A * 4);
constexpr size_t O_RSM  = al(O_RMX + 512);
constexpr size_t O_HEAD = al(O_RSM + 512);
constexpr size_t O_NEXT = al(O_HEAD + (size_t)BB * T_INFO * 4);
constexpr size_t O_QT   = al(O_NEXT + (size_t)BB * NSENT * 4);

extern "C" void kernel_launch(void* const* d_in, const int* in_sizes, int n_in,
                              void* d_out, int out_size, void* d_ws, size_t ws_size,
                              hipStream_t stream) {
  (void)in_sizes; (void)n_in; (void)out_size; (void)ws_size;
  const int*   info      = (const int*)d_in[0];
  const int*   info_idx  = (const int*)d_in[1];
  const int*   num_sent  = (const int*)d_in[2];
  const int*   question  = (const int*)d_in[3];
  const int*   qidx      = (const int*)d_in[4];
  const float* info_emb  = (const float*)d_in[5];
  const float* Wi = (const float*)d_in[6];
  const float* Ui = (const float*)d_in[7];
  const float* bi = (const float*)d_in[8];
  const float* q_emb = (const float*)d_in[9];
  const float* Wq = (const float*)d_in[10];
  const float* Uq = (const float*)d_in[11];
  const float* bq = (const float*)d_in[12];
  const float* Wm = (const float*)d_in[13];
  const float* Um = (const float*)d_in[14];
  const float* bm = (const float*)d_in[15];
  const float* Wa = (const float*)d_in[16];
  const float* Ua = (const float*)d_in[17];
  const float* ba = (const float*)d_in[18];

  char* ws = (char*)d_ws;
  auto WSP = [&](size_t o) -> void* { return (void*)(ws + o); };
  unsigned* bar = (unsigned*)(ws + O_BAR);

  // barriers + index structures
  k_fill_u32<<<1, 64, 0, stream>>>((uint32_t*)bar, 0u, 64);
  k_fill_u32<<<512, 256, 0, stream>>>((uint32_t*)WSP(O_HEAD), 0xFFFFFFFFu, (size_t)BB * T_INFO);
  k_build_match<<<1, 128, 0, stream>>>(info_idx, T_INFO, (int*)WSP(O_HEAD), (int*)WSP(O_NEXT));
  k_qtgt<<<1, 128, 0, stream>>>(qidx, (int*)WSP(O_QT));

  // weight packing (f32 -> bf16 fragment-major, padded)
  k_pack<<<512,  256, 0, stream>>>(Wi, EMB, RU, KC_E, 3 * NT_R, HP_R, (__bf16*)WSP(O_WIP));
  k_pack<<<1024, 256, 0, stream>>>(Ui, RU,  RU, KC_R, 3 * NT_R, HP_R, (__bf16*)WSP(O_UIP));
  k_pack<<<512,  256, 0, stream>>>(Wq, EMB, RU, KC_E, 3 * NT_R, HP_R, (__bf16*)WSP(O_WQP));
  k_pack<<<1024, 256, 0, stream>>>(Uq, RU,  RU, KC_R, 3 * NT_R, HP_R, (__bf16*)WSP(O_UQP));
  k_pack<<<1024, 256, 0, stream>>>(Wm, RU,  MU, KC_R, 3 * NT_M, HP_M, (__bf16*)WSP(O_WMP));
  k_pack<<<1024, 256, 0, stream>>>(Um, MU,  MU, KC_M, 3 * NT_M, HP_M, (__bf16*)WSP(O_UMP));
  k_pack<<<2048, 256, 0, stream>>>(Wa, MU,  VU, KC_M, 3 * NT_A, HP_A, (__bf16*)WSP(O_WAP));
  k_pack<<<4096, 256, 0, stream>>>(Ua, VU,  VU, KC_A, 3 * NT_A, HP_A, (__bf16*)WSP(O_UAP));

  // embeddings -> padded bf16 x inputs
  k_embed<<<4096, 256, 0, stream>>>(info,     info_emb, T_INFO, EMB, EP, (__bf16*)WSP(O_XI));
  k_embed<<<512,  256, 0, stream>>>(question, q_emb,    T_Q,    EMB, EP, (__bf16*)WSP(O_XQ));

  float*  hf = (float*)WSP(O_HF);
  __bf16* hb = (__bf16*)WSP(O_HB);
  auto zeroH = [&]() {
    k_fill_u32<<<1024, 256, 0, stream>>>((uint32_t*)WSP(O_HF), 0u, (size_t)2 * BB * HP_A);
    k_fill_u32<<<512,  256, 0, stream>>>((uint32_t*)WSP(O_HB), 0u, (size_t)BB * HP_A);
  };

  // ---- info GRU: T=1024, scatter h[t] rows at info_idx into enc_info ----
  zeroH();
  k_gru_scan_tanh<<<NT_R, 256, 0, stream>>>(
      (const __bf16*)WSP(O_XI), T_INFO * EP, KC_E, (const __bf16*)WSP(O_WIP),
      (const __bf16*)WSP(O_UIP), KC_R, HP_R, RU, NT_R,
      bi, bi + 3 * RU, T_INFO, hf, hb,
      0, (float*)WSP(O_EI),
      (const int*)WSP(O_HEAD), (const int*)WSP(O_NEXT), nullptr, nullptr,
      nullptr, bar + 4);

  // ---- question GRU: T=64, scatter h at question_idx-1 into enc_q ----
  zeroH();
  k_gru_scan_tanh<<<NT_R, 256, 0, stream>>>(
      (const __bf16*)WSP(O_XQ), T_Q * EP, KC_E, (const __bf16*)WSP(O_WQP),
      (const __bf16*)WSP(O_UQP), KC_R, HP_R, RU, NT_R,
      bq, bq + 3 * RU, T_Q, hf, hb,
      1, (float*)WSP(O_EQ),
      nullptr, nullptr, (const int*)WSP(O_QT), nullptr,
      nullptr, bar + 8);

  // ---- prod = enc_info * pad * enc_q (bf16 x-input of memory GRU) ----
  k_prod<<<2048, 256, 0, stream>>>((const float*)WSP(O_EI), (const float*)WSP(O_EQ),
                                   num_sent, HP_R, RU, (__bf16*)WSP(O_XM));

  // ---- memory GRU: T=64, masked bf16 sequence output (x-input of answer GRU) ----
  zeroH();
  k_gru_scan_tanh<<<NT_M, 256, 0, stream>>>(
      (const __bf16*)WSP(O_XM), NSENT * HP_R, KC_R, (const __bf16*)WSP(O_WMP),
      (const __bf16*)WSP(O_UMP), KC_M, HP_M, MU, NT_M,
      bm, bm + 3 * MU, NSENT, hf, hb,
      2, nullptr, nullptr, nullptr, nullptr, num_sent,
      (__bf16*)WSP(O_XA), bar + 12);

  // ---- answer GRU: T=64, softmax candidate, writes d_out [B,64,V] ----
  zeroH();
  k_gru_scan_softmax<<<NT_A, 256, 0, stream>>>(
      (const __bf16*)WSP(O_XA), NSENT * HP_M, KC_M, (const __bf16*)WSP(O_WAP),
      (const __bf16*)WSP(O_UAP), KC_A, HP_A, VU, NT_A,
      ba, ba + 3 * VU, NSENT, hf, hb,
      (float*)WSP(O_SB), (float*)WSP(O_ZB), (float*)WSP(O_RMX), (float*)WSP(O_RSM),
      (float*)d_out, bar + 16);
}
